// MultiHeadAttention_76020921139945
// MI455X (gfx1250) — matrix operand order repaired and checked by reading, not compiled, not run on hardware
//
#include <hip/hip_runtime.h>
#include <hip/hip_bf16.h>

typedef __bf16 bf16;
typedef __attribute__((ext_vector_type(16))) __bf16 v16bf;
typedef __attribute__((ext_vector_type(8)))  float  v8f;
typedef int v4i_t __attribute__((vector_size(16)));   // matches builtin param type

#if defined(__has_builtin)
#  if __has_builtin(__builtin_amdgcn_global_load_async_to_lds_b128)
#    define ASYNC_LDS 1
#  endif
#endif
#ifndef ASYNC_LDS
#  define ASYNC_LDS 0
#endif

// 16-byte global->LDS copy: async path on CDNA5 (ASYNCcnt), VGPR round-trip otherwise.
__device__ __forceinline__ void asyncCopy16(const bf16* g, bf16* l) {
#if ASYNC_LDS
  __builtin_amdgcn_global_load_async_to_lds_b128(
      (v4i_t*)(void*)g, (v4i_t*)(void*)l, 0, 0);
#else
  *(uint4*)l = *(const uint4*)g;
#endif
}

__device__ __forceinline__ void waitAsync() {
#if ASYNC_LDS
#  if __has_builtin(__builtin_amdgcn_s_wait_asynccnt)
  __builtin_amdgcn_s_wait_asynccnt(0);
#  else
  asm volatile("s_wait_asynccnt 0x0" ::: "memory");
#  endif
#endif
}

union FragU { v16bf v; unsigned u[8]; };

// A-matrix fragment (16x32 bf16, M=lane&15): VGPR p<4 -> K = 2p+b + 8*half,
// p>=4 -> K = 16 + 2(p-4)+b + 8*half.  Pairs contiguous -> dword loads.
__device__ inline v16bf ldFragA(const bf16* base, int strideEl, int kOff, int lane) {
  const int m = lane & 15, half = lane >> 4;
  const unsigned* p32 = (const unsigned*)base;
  FragU f;
#pragma unroll
  for (int p = 0; p < 8; ++p) {
    int k = kOff + ((p & 4) << 2) + ((p & 3) << 1) + (half << 3);
    f.u[p] = p32[(m * strideEl + k) >> 1];
  }
  return f.v;
}

// B-matrix fragment (32x16 bf16, N=lane&15): VGPR p<4 -> K = 2p+b + 8*half,
// p>=4 -> K = 16 + 2(p-4)+b + 8*half (same order as ldFragA).
__device__ inline v16bf ldFragB(const bf16* base, int strideEl, int kOff, int lane) {
  const int n = lane & 15, half = lane >> 4;
  const unsigned* p32 = (const unsigned*)base;
  FragU f;
#pragma unroll
  for (int p = 0; p < 8; ++p) {
    int k = kOff + ((p & 4) << 2) + ((p & 3) << 1) + (half << 3);
    f.u[p] = p32[(n * strideEl + k) >> 1];
  }
  return f.v;
}

// ---------------------------------------------------------------------------
__global__ void cvt_f32_bf16(const float* __restrict__ s, bf16* __restrict__ d, int n) {
  int i2 = (blockIdx.x * blockDim.x + threadIdx.x) * 2;
  if (i2 < n) {
    float2 f = *(const float2*)(s + i2);
    d[i2]     = (bf16)f.x;
    d[i2 + 1] = (bf16)f.y;
  }
}

// ---------------------------------------------------------------------------
// GEMM: Y[M,N] = (X[M,K] @ W[N,K]^T + bias) * outScale
// mode 0: f32 row-major to outF.  mode 1: bf16 head-separated [B,H,T,hd] to outH.
// Block: 256 thr = 8 waves; tile 64(M) x 128(N); wave tile 16 x 64.
// Double-buffered async LDS staging, one barrier per 32-deep K chunk.
// ---------------------------------------------------------------------------
__global__ __launch_bounds__(256)
void gemm_bf16(const bf16* __restrict__ X, const bf16* __restrict__ W,
               const float* __restrict__ bias, float* __restrict__ outF,
               bf16* __restrict__ outH, int M, int N, int K,
               float outScale, int mode)
{
  const int lane = threadIdx.x & 31, wave = threadIdx.x >> 5;
  const int waveM = wave >> 1, waveN = wave & 1;
  const int blockN = blockIdx.x * 128, blockM = blockIdx.y * 64;

  __shared__ bf16 As[2][64][40];   // 80B rows: 16B-aligned, bank-spread
  __shared__ bf16 Bs[2][128][40];

  v8f acc[4] = {};
  const int arow = threadIdx.x >> 2, aseg = threadIdx.x & 3;

  auto stage = [&](int buf, int k0) {
    asyncCopy16(X + (size_t)(blockM + arow) * K + k0 + aseg * 8,
                &As[buf][arow][aseg * 8]);
#pragma unroll
    for (int it = 0; it < 2; ++it) {
      int r = arow + it * 64;
      asyncCopy16(W + (size_t)(blockN + r) * K + k0 + aseg * 8,
                  &Bs[buf][r][aseg * 8]);
    }
  };

  stage(0, 0);
  waitAsync();
  __syncthreads();

  for (int k0 = 0; k0 < K; k0 += 32) {
    const int buf = (k0 >> 5) & 1;
    const bool more = (k0 + 32) < K;
    if (more) stage(buf ^ 1, k0 + 32);   // buf^1 free since end-of-previous barrier

    v16bf a = ldFragA(&As[buf][waveM * 16][0], 40, 0, lane);
    v16bf bfr[4];
#pragma unroll
    for (int j = 0; j < 4; ++j)
      bfr[j] = ldFragB(&Bs[buf][waveN * 64 + j * 16][0], 40, 0, lane);
#pragma unroll
    for (int j = 0; j < 4; ++j)
      acc[j] = __builtin_amdgcn_wmma_f32_16x16x32_bf16(
          false, a, false, bfr[j], (short)0, acc[j], false, false);

    if (more) { waitAsync(); __syncthreads(); }
  }

  const int half = lane >> 4, nl = lane & 15;
  const int m0 = blockM + waveM * 16, n0 = blockN + waveN * 64;
#pragma unroll
  for (int j = 0; j < 4; ++j) {
    int n = n0 + j * 16 + nl;
    float bv = bias[n];
#pragma unroll
    for (int i = 0; i < 8; ++i) {
      int m = m0 + i + 8 * half;                 // C/D layout: M = vgpr + 8*(lane/16)
      float y = (acc[j][i] + bv) * outScale;
      if (mode == 0) {
        outF[(size_t)m * N + n] = y;
      } else {                                   // [B,H,T,hd], T=2048, hd=64, H=16
        int bb = m >> 11, t = m & 2047, hh = n >> 6, dd = n & 63;
        outH[(((size_t)(bb * 16 + hh)) * 2048 + t) * 64 + dd] = (bf16)y;
      }
    }
  }
}

// ---------------------------------------------------------------------------
// Flash attention over head-major Q/K/V [B*H, T, 64] bf16.
// Q pre-scaled by (1/sqrt(64))*log2(e) -> exp2f softmax.
// Block (256 thr): one (b,h) x 128 q-rows; wave w owns rows w*16..w*16+15.
// ---------------------------------------------------------------------------
__global__ __launch_bounds__(256)
void flash_attn(const bf16* __restrict__ Qh, const bf16* __restrict__ Kh,
                const bf16* __restrict__ Vh, bf16* __restrict__ AO)
{
  const int lane = threadIdx.x & 31, wave = threadIdx.x >> 5;
  const int bh = blockIdx.x;                 // 0..31
  const int qt = blockIdx.y;                 // 0..15
  const int b = bh >> 4, h = bh & 15;
  const size_t hb = (size_t)bh * 2048 * 64;
  const int qrow0 = qt * 128 + wave * 16;
  const int half = lane >> 4, nl = lane & 15;

  __shared__ bf16 Ks[64][72];                // [key][d], 144B rows (async-staged)
  __shared__ bf16 Vt[64][66];                // [d][key] (register-transposed)
  __shared__ bf16 Ps[8][16][68];             // per-wave P scratch (C->A relayout)

  v16bf qf[2];
  qf[0] = ldFragA(Qh + hb + (size_t)qrow0 * 64, 64, 0, lane);
  qf[1] = ldFragA(Qh + hb + (size_t)qrow0 * 64, 64, 32, lane);

  float mrun[8], lrun[8];
  v8f o[4] = {};
#pragma unroll
  for (int i = 0; i < 8; ++i) { mrun[i] = -__builtin_inff(); lrun[i] = 0.f; }

  for (int kb = 0; kb < 2048 / 64; ++kb) {
    const int k0 = kb * 64;
    __syncthreads();                         // tiles free for overwrite
#pragma unroll
    for (int it = 0; it < 2; ++it) {
      int row = (threadIdx.x >> 3) + it * 32;
      int seg = threadIdx.x & 7;
      asyncCopy16(Kh + hb + (size_t)(k0 + row) * 64 + seg * 8, &Ks[row][seg * 8]);
      uint4 tv = *(const uint4*)(Vh + hb + (size_t)(k0 + row) * 64 + seg * 8);
      const bf16* pe = (const bf16*)&tv;
#pragma unroll
      for (int e = 0; e < 8; ++e) Vt[seg * 8 + e][row] = pe[e];
    }
    waitAsync();
    __syncthreads();

    // S = Q K^T : 4 n-tiles of 16 keys, K-dim = 64 (2 WMMAs each)
    v8f st[4];
#pragma unroll
    for (int t = 0; t < 4; ++t) {
      v16bf bk0 = ldFragB(&Ks[t * 16][0], 72, 0, lane);
      v16bf bk1 = ldFragB(&Ks[t * 16][0], 72, 32, lane);
      v8f s = {};
      s = __builtin_amdgcn_wmma_f32_16x16x32_bf16(false, qf[0], false, bk0,
                                                  (short)0, s, false, false);
      s = __builtin_amdgcn_wmma_f32_16x16x32_bf16(false, qf[1], false, bk1,
                                                  (short)0, s, false, false);
      st[t] = s;
    }

    // online softmax (rows live in 16-lane half-groups; xor<=8 stays in group)
    float mnew[8], rsum[8];
#pragma unroll
    for (int i = 0; i < 8; ++i)
      mnew[i] = fmaxf(fmaxf(st[0][i], st[1][i]), fmaxf(st[2][i], st[3][i]));
#pragma unroll
    for (int dx = 1; dx < 16; dx <<= 1)
#pragma unroll
      for (int i = 0; i < 8; ++i)
        mnew[i] = fmaxf(mnew[i], __shfl_xor(mnew[i], dx, 32));
#pragma unroll
    for (int i = 0; i < 8; ++i) { mnew[i] = fmaxf(mnew[i], mrun[i]); rsum[i] = 0.f; }

#pragma unroll
    for (int t = 0; t < 4; ++t)
#pragma unroll
      for (int i = 0; i < 8; ++i) {
        float p = exp2f(st[t][i] - mnew[i]);
        rsum[i] += p;
        Ps[wave][i + 8 * half][t * 16 + nl] = (bf16)p;   // row-major P
      }
#pragma unroll
    for (int dx = 1; dx < 16; dx <<= 1)
#pragma unroll
      for (int i = 0; i < 8; ++i)
        rsum[i] += __shfl_xor(rsum[i], dx, 32);

#pragma unroll
    for (int i = 0; i < 8; ++i) {
      float alpha = exp2f(mrun[i] - mnew[i]);
      lrun[i] = lrun[i] * alpha + rsum[i];
      mrun[i] = mnew[i];
#pragma unroll
      for (int dt = 0; dt < 4; ++dt) o[dt][i] *= alpha;
    }

    asm volatile("" ::: "memory");   // keep per-wave P stores before re-gather

    // O += P V : A = P (16x64 keys), B = V^T tile (keys x d)
#pragma unroll
    for (int kc = 0; kc < 2; ++kc) {
      v16bf pa = ldFragA(&Ps[wave][0][0], 68, kc * 32, lane);
      v16bf bv[4];
#pragma unroll
      for (int dt = 0; dt < 4; ++dt)
        bv[dt] = ldFragB(&Vt[dt * 16][0], 66, kc * 32, lane);
#pragma unroll
      for (int dt = 0; dt < 4; ++dt)
        o[dt] = __builtin_amdgcn_wmma_f32_16x16x32_bf16(
            false, pa, false, bv[dt], (short)0, o[dt], false, false);
    }
  }

  // finalize: O/l, store bf16 [B,T,H*hd]
#pragma unroll
  for (int i = 0; i < 8; ++i) {
    float rinv = 1.f / lrun[i];
    int t = qrow0 + i + 8 * half;
#pragma unroll
    for (int dt = 0; dt < 4; ++dt) {
      int d = dt * 16 + nl;
      AO[((size_t)(b * 2048 + t)) * 1024 + h * 64 + d] = (bf16)(o[dt][i] * rinv);
    }
  }
}

// ---------------------------------------------------------------------------
extern "C" void kernel_launch(void* const* d_in, const int* in_sizes, int n_in,
                              void* d_out, int out_size, void* d_ws, size_t ws_size,
                              hipStream_t stream) {
  (void)in_sizes; (void)n_in; (void)out_size; (void)ws_size;
  const float* q  = (const float*)d_in[0];
  const float* k  = (const float*)d_in[1];
  const float* v  = (const float*)d_in[2];
  const float* Wq = (const float*)d_in[3];
  const float* bq = (const float*)d_in[4];
  const float* Wk = (const float*)d_in[5];
  const float* bk = (const float*)d_in[6];
  const float* Wv = (const float*)d_in[7];
  const float* bv = (const float*)d_in[8];
  const float* Wo = (const float*)d_in[9];
  const float* bo = (const float*)d_in[10];
  float* out = (float*)d_out;

  bf16* ws = (bf16*)d_ws;
  const size_t MT = 4096ull * 1024ull;   // activation elements
  const size_t KT = 1024ull * 1024ull;   // weight elements
  bf16 *qb = ws,      *kb = qb + MT, *vb = kb + MT;
  bf16 *wq = vb + MT, *wk = wq + KT, *wv = wk + KT, *wo = wv + KT;
  bf16 *qh = wo + KT, *kh = qh + MT, *vh = kh + MT, *ao = vh + MT;

  cvt_f32_bf16<<<(unsigned)(MT / 512), 256, 0, stream>>>(q,  qb, (int)MT);
  cvt_f32_bf16<<<(unsigned)(MT / 512), 256, 0, stream>>>(k,  kb, (int)MT);
  cvt_f32_bf16<<<(unsigned)(MT / 512), 256, 0, stream>>>(v,  vb, (int)MT);
  cvt_f32_bf16<<<(unsigned)(KT / 512), 256, 0, stream>>>(Wq, wq, (int)KT);
  cvt_f32_bf16<<<(unsigned)(KT / 512), 256, 0, stream>>>(Wk, wk, (int)KT);
  cvt_f32_bf16<<<(unsigned)(KT / 512), 256, 0, stream>>>(Wv, wv, (int)KT);
  cvt_f32_bf16<<<(unsigned)(KT / 512), 256, 0, stream>>>(Wo, wo, (int)KT);

  dim3 gg(1024 / 128, 4096 / 64);
  const float qscale = 0.125f * 1.44269504088896340736f;  // 1/sqrt(hd) * log2(e)
  gemm_bf16<<<gg, 256, 0, stream>>>(qb, wq, bq, nullptr, qh, 4096, 1024, 1024, qscale, 1);
  gemm_bf16<<<gg, 256, 0, stream>>>(kb, wk, bk, nullptr, kh, 4096, 1024, 1024, 1.0f, 1);
  gemm_bf16<<<gg, 256, 0, stream>>>(vb, wv, bv, nullptr, vh, 4096, 1024, 1024, 1.0f, 1);

  flash_attn<<<dim3(32, 16), 256, 0, stream>>>(qh, kh, vh, ao);

  gemm_bf16<<<gg, 256, 0, stream>>>(ao, wo, bo, out, nullptr, 4096, 1024, 1024, 1.0f, 0);
}
